// EmbeddingNet_25804163514791
// MI455X (gfx1250) — matrix operand order repaired and checked by reading, not compiled, not run on hardware
//
#include <hip/hip_runtime.h>
#include <math.h>

// ---------------- CDNA5 WMMA types ----------------
typedef __attribute__((ext_vector_type(16))) _Float16 v16h;
typedef __attribute__((ext_vector_type(2)))  _Float16 v2h;
typedef __attribute__((ext_vector_type(8)))  float    v8f;

#define TSTEPS 300
#define BATCH  2048
#define DIN    128   // D
#define CEMB   64    // C
#define HP     112   // hidden 100 padded to 7 N-tiles of 16
#define KH     128   // hidden K padded to 4 chunks of 32
#define DOUT   128   // integrand output width = D
#define WAVES  8

#define WMMA_F16(a, b, c) \
  __builtin_amdgcn_wmma_f32_16x16x32_f16(false, (a), false, (b), (short)0, (c), false, false)

union AFrag { v16h v; v2h p[8]; _Float16 h[16]; };
union CFrag { v8f  v; float f[8]; };

// A fragment (16x32 f16, M x K) from row-major LDS buffer buf[m*ld + k].
// Layout used here (contiguous k): lane m = lane&15; half = lane>>4;
//   vgpr r : K = 2r + 16*half  (pairs; contiguous, matches load_b_frag)
__device__ __forceinline__ v16h load_a_frag(const _Float16* buf, int ld, int lane, int kbase) {
  const int m = lane & 15, half = lane >> 4;
  AFrag u;
#pragma unroll
  for (int r = 0; r < 8; ++r) {
    const int k = kbase + 2 * r + 16 * half;
    u.p[r] = *(const v2h*)(buf + m * ld + k);
  }
  return u.v;
}

// B fragment (32x16 f16, K x N) from N-major (transposed) LDS weights w[n*ldk + k].
// Layout used here (contiguous k): n = lane&15 (+16*ntile); lanes 0-15 hold K=kbase..+15, lanes 16-31 K=+16.
// Contiguous in k -> 32-byte LDS read (2x ds_load_b128).
__device__ __forceinline__ v16h load_b_frag(const _Float16* w, int ldk, int lane, int ntile, int kbase) {
  const int n  = (ntile << 4) + (lane & 15);
  const int k0 = kbase + ((lane >> 4) << 4);
  return *(const v16h*)(w + n * ldk + k0);
}

// hidden layer: A(16x128 LDS f16) x W(128x112 padded, N-major LDS) + bias, ReLU -> O(16x128)
__device__ __forceinline__ void hidden_layer(const _Float16* A, const _Float16* W,
                                             const float* bias, _Float16* O, int lane) {
  const int m = lane & 15, half = lane >> 4;
  v16h af[4];
#pragma unroll
  for (int kc = 0; kc < 4; ++kc) af[kc] = load_a_frag(A, KH, lane, kc * 32);
#pragma unroll
  for (int j = 0; j < 7; ++j) {
    CFrag acc;
#pragma unroll
    for (int v = 0; v < 8; ++v) acc.f[v] = 0.f;
#pragma unroll
    for (int kc = 0; kc < 4; ++kc) {
      v16h b = load_b_frag(W, KH, lane, j, kc * 32);
      acc.v = WMMA_F16(af[kc], b, acc.v);
    }
    const int n = (j << 4) + m;
    const float bi = bias[n];
#pragma unroll
    for (int v = 0; v < 8; ++v) {
      float val = acc.f[v] + bi;
      O[(v + 8 * half) * KH + n] = (_Float16)(val > 0.f ? val : 0.f);
    }
  }
  asm volatile("s_wait_dscnt 0" ::: "memory");  // intra-wave LDS producer->consumer
}

// ---------------- kernel 0: weight convert/transpose/pad to f16 (U2,U3,U4) ----------------
__global__ void prep_weights(const float* __restrict__ U2, const float* __restrict__ U3,
                             const float* __restrict__ U4,
                             _Float16* __restrict__ wt2, _Float16* __restrict__ wt3,
                             _Float16* __restrict__ wt4) {
  int i = blockIdx.x * blockDim.x + threadIdx.x;
  const int N2 = HP * KH, N3 = HP * KH, N4 = DOUT * KH;
  if (i < N2) {                       // U2: (100,100) -> wt2[n][k], pad both to 112/128
    int n = i / KH, k = i - n * KH;
    wt2[i] = (_Float16)((n < 100 && k < 100) ? U2[k * 100 + n] : 0.f);
    return;
  }
  i -= N2;
  if (i < N3) {
    int n = i / KH, k = i - n * KH;
    wt3[i] = (_Float16)((n < 100 && k < 100) ? U3[k * 100 + n] : 0.f);
    return;
  }
  i -= N3;
  if (i < N4) {                       // U4: (100,128) -> wt4[n][k], k padded
    int n = i / KH, k = i - n * KH;
    wt4[i] = (_Float16)((k < 100) ? U4[k * DOUT + n] : 0.f);
  }
}

// ---------------- kernel 1: embedding net + t-independent layer-1 precompute ----------------
// h = relu-MLP(x);  P[b,n] = sum_k x[b,k]*U1[k,n]  (k<128)
//                   Qb[b,n] = c1[n] + sum_k h[b,k]*U1[128+k,n]
// Per-t layer 1 then collapses to relu(ts*P + Qb) in umnn_main.
__global__ __launch_bounds__(256) void embed_net(
    const float* __restrict__ xg,
    const float* __restrict__ W1, const float* __restrict__ b1,
    const float* __restrict__ W2, const float* __restrict__ b2,
    const float* __restrict__ W3, const float* __restrict__ b3,
    const float* __restrict__ U1, const float* __restrict__ c1,
    float* __restrict__ Pg, float* __restrict__ Qg) {
  __shared__ float xr[DIN];
  __shared__ float a1[200];
  __shared__ float a2[200];
  __shared__ float hh[CEMB];
  const int b = blockIdx.x, tid = threadIdx.x;
  if (tid < DIN) xr[tid] = xg[b * DIN + tid];
  __syncthreads();
  if (tid < 200) {
    float s = b1[tid];
    for (int k = 0; k < DIN; ++k) s += xr[k] * W1[k * 200 + tid];
    a1[tid] = s > 0.f ? s : 0.f;
  }
  __syncthreads();
  if (tid < 200) {
    float s = b2[tid];
    for (int k = 0; k < 200; ++k) s += a1[k] * W2[k * 200 + tid];
    a2[tid] = s > 0.f ? s : 0.f;
  }
  __syncthreads();
  if (tid < CEMB) {
    float s = b3[tid];
    for (int k = 0; k < 200; ++k) s += a2[k] * W3[k * CEMB + tid];
    hh[tid] = s > 0.f ? s : 0.f;
  }
  __syncthreads();
  if (tid < HP) {                 // n = tid
    float p = 0.f, q = 0.f;
    if (tid < 100) {
      for (int k = 0; k < DIN; ++k) p += xr[k] * U1[k * 100 + tid];
      q = c1[tid];
      for (int k = 0; k < CEMB; ++k) q += hh[k] * U1[(DIN + k) * 100 + tid];
    }
    Pg[b * HP + tid] = p;
    Qg[b * HP + tid] = q;
  }
}

// ---------------- kernel 2: fused UMNN quadrature MLP (WMMA) ----------------
__global__ __launch_bounds__(256, 1) void umnn_main(
    const float* __restrict__ xg,
    const float* __restrict__ Pg, const float* __restrict__ Qg,
    const _Float16* __restrict__ gw2, const _Float16* __restrict__ gw3,
    const _Float16* __restrict__ gw4,
    const float* __restrict__ c2, const float* __restrict__ c3,
    const float* __restrict__ c4,
    float* __restrict__ out) {
  __shared__ __align__(64) _Float16 w2[HP * KH];
  __shared__ __align__(64) _Float16 w3[HP * KH];
  __shared__ __align__(64) _Float16 w4[DOUT * KH];
  __shared__ float bias2[HP], bias3[HP], bias4[DOUT];
  __shared__ __align__(16) float xs[16 * DIN];
  __shared__ __align__(16) float Pl[16 * KH];      // ts-independent layer-1 terms, zero-padded
  __shared__ __align__(16) float Qbl[16 * KH];
  __shared__ float Facc[16];
  __shared__ __align__(16) _Float16 act0[WAVES][16 * KH];
  __shared__ __align__(16) _Float16 act1[WAVES][16 * KH];

  const int tid = threadIdx.x;
  const int b0  = blockIdx.x * 16;

  // ---- one-time stage into LDS ----
  for (int i = tid; i < HP * KH; i += 256) { w2[i] = gw2[i]; w3[i] = gw3[i]; }
  for (int i = tid; i < DOUT * KH; i += 256) w4[i] = gw4[i];
  if (tid < HP) {
    bias2[tid] = (tid < 100) ? c2[tid] : 0.f;
    bias3[tid] = (tid < 100) ? c3[tid] : 0.f;
  }
  if (tid < DOUT) bias4[tid] = c4[tid];
  for (int i = tid; i < 16 * DIN; i += 256) xs[i] = xg[b0 * DIN + i];
  for (int i = tid; i < 16 * KH; i += 256) {       // pad 112 -> 128 with zeros
    const int m = i >> 7, n = i & 127;
    Pl[i]  = (n < HP) ? Pg[(b0 + m) * HP + n] : 0.f;
    Qbl[i] = (n < HP) ? Qg[(b0 + m) * HP + n] : 0.f;
  }
  if (tid < 16) Facc[tid] = 0.f;
  for (int i = tid; i < WAVES * 16 * KH; i += 256) {  // zero pad cols (hidden layers write n<112)
    (&act0[0][0])[i] = (_Float16)0.f;
    (&act1[0][0])[i] = (_Float16)0.f;
  }
  __syncthreads();

  const int w = tid >> 5;
  const int lane = tid & 31;
  const int m = lane & 15, half = lane >> 4;
  _Float16* p0 = act0[w];
  _Float16* p1 = act1[w];

  // wave-uniform quadrature loop: wave w handles t = w, w+8, ... (EXEC all-ones at WMMAs)
  for (int t = w; t < TSTEPS; t += WAVES) {
    const float ts = (t + 0.5f) * (1.0f / TSTEPS);

    // layer 1 via precomputed affine form: p0 = relu(ts*P + Qb), f32 math, packed f16 stores.
    // Uniform trip count (32 iters/lane) -> no divergence.
    for (int i = lane * 2; i < 16 * KH; i += 64) {
      float2 pv = *(const float2*)(Pl + i);
      float2 qv = *(const float2*)(Qbl + i);
      float a0 = ts * pv.x + qv.x; a0 = a0 > 0.f ? a0 : 0.f;
      float a1 = ts * pv.y + qv.y; a1 = a1 > 0.f ? a1 : 0.f;
      v2h o; o[0] = (_Float16)a0; o[1] = (_Float16)a1;
      *(v2h*)(p0 + i) = o;
    }
    asm volatile("s_wait_dscnt 0" ::: "memory");

    hidden_layer(p0, w2, bias2, p1, lane);   // layer 2
    hidden_layer(p1, w3, bias3, p0, lane);   // layer 3

    // layer 4: (16x128) x (128x128), ELU+1, dot with x, quadrature accumulate
    v16h af4[4];
#pragma unroll
    for (int kc = 0; kc < 4; ++kc) af4[kc] = load_a_frag(p0, KH, lane, kc * 32);
    float fs[8];
#pragma unroll
    for (int v = 0; v < 8; ++v) fs[v] = 0.f;
#pragma unroll
    for (int j = 0; j < 8; ++j) {
      CFrag acc;
#pragma unroll
      for (int v = 0; v < 8; ++v) acc.f[v] = 0.f;
#pragma unroll
      for (int kc = 0; kc < 4; ++kc) {
        v16h b = load_b_frag(w4, KH, lane, j, kc * 32);
        acc.v = WMMA_F16(af4[kc], b, acc.v);
      }
      const int n = (j << 4) + m;           // n == output dim d
      const float bi = bias4[n];
#pragma unroll
      for (int v = 0; v < 8; ++v) {
        float val = acc.f[v] + bi;
        float f = (val > 0.f) ? (val + 1.f) : __expf(val);   // elu(x)+1
        fs[v] += f * xs[(v + 8 * half) * DIN + n];
      }
    }
    // reduce over the 16 lanes (16 d-columns) of each half, then atomic per batch row
#pragma unroll
    for (int v = 0; v < 8; ++v) {
      float s = fs[v];
      s += __shfl_xor(s, 1, 32);
      s += __shfl_xor(s, 2, 32);
      s += __shfl_xor(s, 4, 32);
      s += __shfl_xor(s, 8, 32);
      if (m == 0) atomicAdd(&Facc[v + 8 * half], s);
    }
  }

  asm volatile("s_wait_dscnt 0" ::: "memory");
  __syncthreads();
  if (tid < 16) {
    float F = Facc[tid] * (1.0f / TSTEPS);
    out[b0 + tid] = 1.0f / (1.0f + __expf(-F));
  }
}

// ---------------- host launcher ----------------
extern "C" void kernel_launch(void* const* d_in, const int* in_sizes, int n_in,
                              void* d_out, int out_size, void* d_ws, size_t ws_size,
                              hipStream_t stream) {
  (void)in_sizes; (void)n_in; (void)out_size; (void)ws_size;
  const float* x  = (const float*)d_in[0];
  const float* W1 = (const float*)d_in[1];
  const float* b1 = (const float*)d_in[2];
  const float* W2 = (const float*)d_in[3];
  const float* b2 = (const float*)d_in[4];
  const float* W3 = (const float*)d_in[5];
  const float* b3 = (const float*)d_in[6];
  const float* U1 = (const float*)d_in[7];
  const float* c1 = (const float*)d_in[8];
  const float* U2 = (const float*)d_in[9];
  const float* c2 = (const float*)d_in[10];
  const float* U3 = (const float*)d_in[11];
  const float* c3 = (const float*)d_in[12];
  const float* U4 = (const float*)d_in[13];
  const float* c4 = (const float*)d_in[14];

  char* ws = (char*)d_ws;                  // all offsets 256B-aligned
  float*    Pg  = (float*)(ws + 0);               // 2048*112*4 = 917504
  float*    Qg  = (float*)(ws + 917504);          // 2048*112*4 = 917504
  _Float16* wt2 = (_Float16*)(ws + 1835008);      // 112*128*2  =  28672
  _Float16* wt3 = (_Float16*)(ws + 1863680);      // 112*128*2  =  28672
  _Float16* wt4 = (_Float16*)(ws + 1892352);      // 128*128*2  =  32768

  const int prep_elems = 2 * HP * KH + DOUT * KH;   // 45056
  prep_weights<<<(prep_elems + 255) / 256, 256, 0, stream>>>(U2, U3, U4, wt2, wt3, wt4);
  embed_net<<<BATCH, 256, 0, stream>>>(x, W1, b1, W2, b2, W3, b3, U1, c1, Pg, Qg);
  umnn_main<<<BATCH / 16, 256, 0, stream>>>(x, Pg, Qg, wt2, wt3, wt4, c2, c3, c4,
                                            (float*)d_out);
}
